// ReliabilityGatedCrossAttention_56461640073246
// MI455X (gfx1250) — matrix operand order repaired and checked by reading, not compiled, not run on hardware
//
#include <hip/hip_runtime.h>

// ---------------------------------------------------------------------------
// Types
// ---------------------------------------------------------------------------
typedef __attribute__((ext_vector_type(16))) __bf16        v16bf;
typedef __attribute__((ext_vector_type(8)))  float         v8f;
typedef __attribute__((ext_vector_type(4)))  unsigned int  u32x4;
typedef __attribute__((ext_vector_type(4)))  int           i32x4;
typedef __attribute__((ext_vector_type(4)))  float         f32x4;
typedef __attribute__((ext_vector_type(4)))  unsigned short u16x4;

union Frag {              // one WMMA 16-bit A/B operand (8 VGPRs)
    v16bf        bf;
    u32x4        q[2];
    unsigned int u[8];
};

#define DIMC   1024
#define NROWS  4096      // B * N
#define HDIM   64

// Async global->LDS DMA path (gfx1250), guarded for toolchain portability.
#if defined(__has_builtin)
#if __has_builtin(__builtin_amdgcn_global_load_async_to_lds_b128)
#define HAVE_ASYNC_LDS 1
#endif
#endif
#ifndef HAVE_ASYNC_LDS
#define HAVE_ASYNC_LDS 0
#endif

#if HAVE_ASYNC_LDS
// Builtin signature (from toolchain diagnostic): (int4 AS1*, int4 AS3*, Ii, Ii)
typedef __attribute__((address_space(1))) i32x4 glb_i32x4;
typedef __attribute__((address_space(3))) i32x4 lds_i32x4;
__device__ __forceinline__ void async_wait0() {
#if __has_builtin(__builtin_amdgcn_s_wait_asynccnt)
    __builtin_amdgcn_s_wait_asynccnt(0);
#else
    asm volatile("s_wait_asynccnt 0x0" ::: "memory");
#endif
}
#endif

// float -> bf16, round to nearest even
__device__ __forceinline__ unsigned short f2bf(float f) {
    unsigned int u = __float_as_uint(f);
    unsigned int r = u + 0x7FFFu + ((u >> 16) & 1u);
    return (unsigned short)(r >> 16);
}

__device__ __forceinline__ float swap16f(float v) {       // lane <-> lane^16
    return __int_as_float(__builtin_amdgcn_ds_swizzle(__float_as_int(v), 0x401F));
}
__device__ __forceinline__ float lane_bcast(float v, int src) {
    return __int_as_float(__builtin_amdgcn_ds_bpermute(src << 2, __float_as_int(v)));
}

// ---------------------------------------------------------------------------
// f32 -> bf16 conversion (4 elems / thread)
// ---------------------------------------------------------------------------
__global__ void cvt_f32_bf16(const float* __restrict__ src,
                             unsigned short* __restrict__ dst, int n4) {
    int i = blockIdx.x * 256 + threadIdx.x;
    if (i < n4) {
        f32x4 v = *(const f32x4*)(src + i * 4);
        u16x4 o;
        o[0] = f2bf(v[0]); o[1] = f2bf(v[1]); o[2] = f2bf(v[2]); o[3] = f2bf(v[3]);
        *(u16x4*)(dst + i * 4) = o;
    }
}

__global__ void logr_kernel(const float* __restrict__ rel,
                            float* __restrict__ lr, int n) {
    int i = blockIdx.x * 256 + threadIdx.x;
    if (i < n) lr[i] = __logf(fmaxf(rel[i], 1e-6f));
}

// ---------------------------------------------------------------------------
// GEMM: C[4096,1024] = A[4096,1024] @ W[1024,1024]^T + bias, epilogue scale.
// 256 thr = 8 waves, block tile 128x128, K-step 32. bf16 WMMA, f32 acc.
// Tile staging uses GLOBAL_LOAD_ASYNC_TO_LDS_B128 when available.
// ---------------------------------------------------------------------------
#define LSTR 40                              // LDS row stride in halves (80B)

__global__ void __launch_bounds__(256)
gemm_wmma_bf16(const unsigned short* __restrict__ A,
               const unsigned short* __restrict__ W,
               const float* __restrict__ bias,
               const float* __restrict__ rowScale,   // nullptr -> use cscale
               float cscale,
               unsigned short* __restrict__ outBf,
               float* __restrict__ outF) {
    __shared__ __align__(16) unsigned short lAs[128 * LSTR];
    __shared__ __align__(16) unsigned short lWs[128 * LSTR];

    const int tid  = threadIdx.x;
    const int lane = tid & 31;
    const int wv   = tid >> 5;
    const int ll   = lane & 15;
    const int hl   = lane >> 4;
    const int wm   = (wv & 3) * 32;
    const int wn   = (wv >> 2) * 64;
    const int m0   = blockIdx.x * 128;
    const int n0   = blockIdx.y * 128;

    v8f acc[2][4];
#pragma unroll
    for (int mt = 0; mt < 2; ++mt)
#pragma unroll
        for (int nt = 0; nt < 4; ++nt) acc[mt][nt] = (v8f){0,0,0,0,0,0,0,0};

    for (int k0 = 0; k0 < DIMC; k0 += 32) {
        __syncthreads();
#if HAVE_ASYNC_LDS
#pragma unroll
        for (int i = 0; i < 2; ++i) {                 // 512 16B chunks / 256 thr
            int c   = tid + i * 256;
            int row = c >> 2;
            int co  = (c & 3) * 8;                    // halves
            __builtin_amdgcn_global_load_async_to_lds_b128(
                (glb_i32x4*)&A[(m0 + row) * DIMC + k0 + co],
                (lds_i32x4*)&lAs[row * LSTR + co], 0, 0);
            __builtin_amdgcn_global_load_async_to_lds_b128(
                (glb_i32x4*)&W[(n0 + row) * DIMC + k0 + co],
                (lds_i32x4*)&lWs[row * LSTR + co], 0, 0);
        }
        async_wait0();
#else
#pragma unroll
        for (int i = 0; i < 2; ++i) {
            int c   = tid + i * 256;
            int row = c >> 2;
            int co  = (c & 3) * 8;
            *(u32x4*)&lAs[row * LSTR + co] = *(const u32x4*)&A[(m0 + row) * DIMC + k0 + co];
            *(u32x4*)&lWs[row * LSTR + co] = *(const u32x4*)&W[(n0 + row) * DIMC + k0 + co];
        }
#endif
        __syncthreads();

        Frag a[2], b[4];
#pragma unroll
        for (int mt = 0; mt < 2; ++mt) {              // A frag: 16(M)x32(K)
            int row = wm + mt * 16 + ll;
            int bse = hl * 8;
            a[mt].q[0] = *(const u32x4*)&lAs[row * LSTR + bse];
            a[mt].q[1] = *(const u32x4*)&lAs[row * LSTR + bse + 16];
        }
#pragma unroll
        for (int nt = 0; nt < 4; ++nt) {              // B frag: 32(K)x16(N)
            int row = wn + nt * 16 + ll;
            int bse = hl * 8;
            b[nt].q[0] = *(const u32x4*)&lWs[row * LSTR + bse];
            b[nt].q[1] = *(const u32x4*)&lWs[row * LSTR + bse + 16];
        }
#pragma unroll
        for (int mt = 0; mt < 2; ++mt)
#pragma unroll
            for (int nt = 0; nt < 4; ++nt)
                acc[mt][nt] = __builtin_amdgcn_wmma_f32_16x16x32_bf16(
                    false, a[mt].bf, false, b[nt].bf, (short)0, acc[mt][nt], false, false);
    }

    // epilogue
#pragma unroll
    for (int mt = 0; mt < 2; ++mt)
#pragma unroll
        for (int nt = 0; nt < 4; ++nt) {
            int gcol = n0 + wn + nt * 16 + ll;
            float bv = bias[gcol];
#pragma unroll
            for (int r = 0; r < 8; ++r) {
                int grow  = m0 + wm + mt * 16 + hl * 8 + r;
                float val = acc[mt][nt][r] + bv;
                val *= rowScale ? rowScale[grow] : cscale;
                if (outF) outF[grow * DIMC + gcol] = val;
                else      outBf[grow * DIMC + gcol] = f2bf(val);
            }
        }
}

// ---------------------------------------------------------------------------
// Flash attention. Grid: (NQ/64, B*HEADS). 128 thr = 4 waves; each wave owns
// 16 q rows. Computes S^T = K·Q^T per 32-kv step (softmax row == lane column),
// online softmax, O += P·V with V^T staged in LDS.
// ---------------------------------------------------------------------------
__global__ void __launch_bounds__(128)
attn_wmma(const unsigned short* __restrict__ Q,
          const unsigned short* __restrict__ Km,
          const unsigned short* __restrict__ Vm,
          const float* __restrict__ logr,
          unsigned short* __restrict__ O) {
    __shared__ __align__(16) unsigned short lVt[HDIM * LSTR];   // V^T: [d][kv]

    const int tid  = threadIdx.x;
    const int lane = tid & 31;
    const int wv   = tid >> 5;
    const int ll   = lane & 15;
    const int hl   = lane >> 4;
    const int bh   = blockIdx.y;
    const int b    = bh >> 4;
    const int h    = bh & 15;
    const int hc   = h * HDIM;
    const int rb   = b * 1024;                 // row base for this batch
    const int q0   = blockIdx.x * 64 + wv * 16;

    // Q B-fragments (q pre-scaled by 1/8 at projection): lane = q column
    Frag qf[2];
    {
        int qrow = rb + q0 + ll;
#pragma unroll
        for (int ks = 0; ks < 2; ++ks) {
            int cb = hc + ks * 32 + hl * 8;
            qf[ks].q[0] = *(const u32x4*)&Q[qrow * DIMC + cb];
            qf[ks].q[1] = *(const u32x4*)&Q[qrow * DIMC + cb + 16];
        }
    }

    v8f oacc[4];
#pragma unroll
    for (int dt = 0; dt < 4; ++dt) oacc[dt] = (v8f){0,0,0,0,0,0,0,0};
    float m = -1e30f, l = 0.0f;

    for (int kv0 = 0; kv0 < 1024; kv0 += 32) {
        // ---- stage V^T tile (32 kv x 64 d) into LDS, transpose on store ----
        __syncthreads();
        {
            int r  = tid >> 2;                  // kv row 0..31
            int d0 = (tid & 3) * 16;
            const unsigned short* src = &Vm[(rb + kv0 + r) * DIMC + hc + d0];
            unsigned short tmp[16];
            *(u32x4*)&tmp[0] = *(const u32x4*)src;
            *(u32x4*)&tmp[8] = *(const u32x4*)(src + 8);
#pragma unroll
            for (int i = 0; i < 16; ++i) lVt[(d0 + i) * LSTR + r] = tmp[i];
        }
        __syncthreads();

        // ---- S^T tiles: two 16(kv) x 16(q) fragments ----
        v8f s[2];
#pragma unroll
        for (int t = 0; t < 2; ++t) {
            s[t] = (v8f){0,0,0,0,0,0,0,0};
            int kvrow = rb + kv0 + t * 16 + ll;           // A frag: kv row
#pragma unroll
            for (int ks = 0; ks < 2; ++ks) {
                Frag kf;
                int cb = hc + ks * 32 + hl * 8;
                kf.q[0] = *(const u32x4*)&Km[kvrow * DIMC + cb];
                kf.q[1] = *(const u32x4*)&Km[kvrow * DIMC + cb + 16];
                s[t] = __builtin_amdgcn_wmma_f32_16x16x32_bf16(
                    false, kf.bf, false, qf[ks].bf, (short)0, s[t], false, false);
            }
            // + log(reliability): element r holds kv = kv0 + t*16 + hl*8 + r
            const float* lr = &logr[b * 1024 + kv0 + t * 16 + hl * 8];
            f32x4 l0 = *(const f32x4*)lr;
            f32x4 l1 = *(const f32x4*)(lr + 4);
#pragma unroll
            for (int i = 0; i < 4; ++i) { s[t][i] += l0[i]; s[t][i + 4] += l1[i]; }
        }

        // ---- online softmax (per q column = per lane, merge lane halves) ----
        float tmax = -1e30f;
#pragma unroll
        for (int t = 0; t < 2; ++t)
#pragma unroll
            for (int i = 0; i < 8; ++i) tmax = fmaxf(tmax, s[t][i]);
        tmax = fmaxf(tmax, swap16f(tmax));
        float mnew = fmaxf(m, tmax);

        float rsum = 0.0f;
        Frag pf;                       // P as bf16 A-fragment (16 q x 32 kv)
#pragma unroll
        for (int t = 0; t < 2; ++t)
#pragma unroll
            for (int i = 0; i < 4; ++i) {
                float p0 = __expf(s[t][2 * i]     - mnew);
                float p1 = __expf(s[t][2 * i + 1] - mnew);
                rsum += p0 + p1;
                pf.u[t * 4 + i] = (unsigned int)f2bf(p0) |
                                  ((unsigned int)f2bf(p1) << 16);
            }
        rsum += swap16f(rsum);

        float alpha = __expf(m - mnew);
        l = l * alpha + rsum;
        m = mnew;

        // rescale O accumulators: O-frag row r maps to q = r + hl*8
        float ar[8];
#pragma unroll
        for (int r = 0; r < 8; ++r) ar[r] = lane_bcast(alpha, r + hl * 8);
#pragma unroll
        for (int dt = 0; dt < 4; ++dt)
#pragma unroll
            for (int r = 0; r < 8; ++r) oacc[dt][r] *= ar[r];

        // ---- O += P @ V ----
#pragma unroll
        for (int dt = 0; dt < 4; ++dt) {
            Frag vf;                                // B frag: 32(kv) x 16(d)
            int drow = dt * 16 + ll;
            int bse  = hl * 8;
            vf.q[0] = *(const u32x4*)&lVt[drow * LSTR + bse];
            vf.q[1] = *(const u32x4*)&lVt[drow * LSTR + bse + 16];
            oacc[dt] = __builtin_amdgcn_wmma_f32_16x16x32_bf16(
                false, pf.bf, false, vf.bf, (short)0, oacc[dt], false, false);
        }
    }

    // ---- finalize: O /= l, store bf16 ----
    float li[8];
#pragma unroll
    for (int r = 0; r < 8; ++r) li[r] = 1.0f / lane_bcast(l, r + hl * 8);
#pragma unroll
    for (int dt = 0; dt < 4; ++dt)
#pragma unroll
        for (int r = 0; r < 8; ++r) {
            int grow = rb + q0 + hl * 8 + r;
            int gcol = hc + dt * 16 + ll;
            O[grow * DIMC + gcol] = f2bf(oacc[dt][r] * li[r]);
        }
}

// ---------------------------------------------------------------------------
// Host launcher
// ---------------------------------------------------------------------------
extern "C" void kernel_launch(void* const* d_in, const int* in_sizes, int n_in,
                              void* d_out, int out_size, void* d_ws, size_t ws_size,
                              hipStream_t stream) {
    const float* query  = (const float*)d_in[0];
    const float* kvfeat = (const float*)d_in[1];
    const float* rel    = (const float*)d_in[2];
    const float* Wq     = (const float*)d_in[3];
    const float* bq     = (const float*)d_in[4];
    const float* Wk     = (const float*)d_in[5];
    const float* bk     = (const float*)d_in[6];
    const float* Wv     = (const float*)d_in[7];
    const float* bv     = (const float*)d_in[8];
    const float* Wo     = (const float*)d_in[9];
    const float* bo     = (const float*)d_in[10];

    char* ws = (char*)d_ws;
    const size_t MB = 1u << 20;
    unsigned short* Xq  = (unsigned short*)(ws +  0 * MB);   // 8 MB
    unsigned short* Xkv = (unsigned short*)(ws +  8 * MB);   // 8 MB
    unsigned short* Wqb = (unsigned short*)(ws + 16 * MB);   // 2 MB
    unsigned short* Wkb = (unsigned short*)(ws + 18 * MB);   // 2 MB
    unsigned short* Wvb = (unsigned short*)(ws + 20 * MB);   // 2 MB
    unsigned short* Wob = (unsigned short*)(ws + 22 * MB);   // 2 MB
    unsigned short* Qb  = (unsigned short*)(ws + 24 * MB);   // 8 MB
    unsigned short* Kb  = (unsigned short*)(ws + 32 * MB);   // 8 MB
    unsigned short* Vb  = (unsigned short*)(ws + 40 * MB);   // 8 MB
    unsigned short* Ob  = (unsigned short*)(ws + 48 * MB);   // 8 MB
    float*          Lr  = (float*)         (ws + 56 * MB);   // 16 KB

    // conversions
    const int nBig4 = NROWS * DIMC / 4;      // 1,048,576
    const int nW4   = DIMC * DIMC / 4;       //   262,144
    cvt_f32_bf16<<<nBig4 / 256, 256, 0, stream>>>(query,  Xq,  nBig4);
    cvt_f32_bf16<<<nBig4 / 256, 256, 0, stream>>>(kvfeat, Xkv, nBig4);
    cvt_f32_bf16<<<nW4   / 256, 256, 0, stream>>>(Wq, Wqb, nW4);
    cvt_f32_bf16<<<nW4   / 256, 256, 0, stream>>>(Wk, Wkb, nW4);
    cvt_f32_bf16<<<nW4   / 256, 256, 0, stream>>>(Wv, Wvb, nW4);
    cvt_f32_bf16<<<nW4   / 256, 256, 0, stream>>>(Wo, Wob, nW4);
    logr_kernel<<<16, 256, 0, stream>>>(rel, Lr, 4096);

    // projections (Q folds softmax scale; K/V fold reliability row-scale)
    dim3 gg(NROWS / 128, DIMC / 128);
    gemm_wmma_bf16<<<gg, 256, 0, stream>>>(Xq,  Wqb, bq, nullptr, 0.125f, Qb, nullptr);
    gemm_wmma_bf16<<<gg, 256, 0, stream>>>(Xkv, Wkb, bk, rel,     1.0f,   Kb, nullptr);
    gemm_wmma_bf16<<<gg, 256, 0, stream>>>(Xkv, Wvb, bv, rel,     1.0f,   Vb, nullptr);

    // flash attention
    attn_wmma<<<dim3(1024 / 64, 4 * 16), 128, 0, stream>>>(Qb, Kb, Vb, Lr, Ob);

    // output projection, f32 result
    gemm_wmma_bf16<<<gg, 256, 0, stream>>>(Ob, Wob, bo, nullptr, 1.0f, nullptr, (float*)d_out);
}
